// Encoder_69295002353761
// MI455X (gfx1250) — compile-verified, matrix operand order checked by reading, not run on hardware
//
#include <hip/hip_runtime.h>
#include <hip/hip_bf16.h>

typedef unsigned short u16;
typedef __attribute__((ext_vector_type(16))) __bf16        v16bf;
typedef __attribute__((ext_vector_type(8)))  float         v8f;
typedef __attribute__((ext_vector_type(4)))  unsigned int  v4u;

#define NN    100000
#define EE    1600000
#define RR    8
#define HIDD  64
#define GG    64
#define NPAD  100032   // 64-row padded node count

// ---------- helpers ----------
__device__ inline u16 f2bf(float f) {           // round-to-nearest-even f32 -> bf16
  unsigned u = __float_as_uint(f);
  unsigned r = u + 0x7FFFu + ((u >> 16) & 1u);
  return (u16)(r >> 16);
}
__device__ inline unsigned encf(float f) {      // order-preserving f32 -> u32
  unsigned u = __float_as_uint(f);
  return (u & 0x80000000u) ? ~u : (u | 0x80000000u);
}
__device__ inline float decf(unsigned u) {
  unsigned v = (u & 0x80000000u) ? (u & 0x7FFFFFFFu) : ~u;
  return __uint_as_float(v);
}

// ---------- 0: combine bases -> transposed bf16 weights  Wt[rel][o][i] ----------
// rel 0..7 : sum_b comp[r,b] * bases[b,i,o];  rel 8 : root[i,o];  i >= din -> 0
__global__ void k_wcombine(const float* __restrict__ comp,
                           const float* __restrict__ bases,
                           const float* __restrict__ root,
                           u16* __restrict__ wt, int din, int kpad) {
  int idx = blockIdx.x * 256 + threadIdx.x;      // [r][o][i], 9*64*kpad total
  int i = idx % kpad;
  int o = (idx / kpad) & 63;
  int r = idx / (kpad * 64);
  float v = 0.f;
  if (i < din) {
    if (r < RR) {
      #pragma unroll
      for (int b = 0; b < 8; ++b)
        v += comp[r * 8 + b] * bases[((size_t)b * din + i) * 64 + o];
    } else {
      v = root[(size_t)i * 64 + o];
    }
  }
  wt[idx] = f2bf(v);
}

// ---------- 1: embedding concat -> bf16 x0 (row stride 128, zero padded) ----------
__global__ void k_embed(const int* __restrict__ x_type, const int* __restrict__ x_sub,
                        const float* __restrict__ x_bow, const float* __restrict__ x_hot,
                        const float* __restrict__ x_topo,
                        const float* __restrict__ emb_type, const float* __restrict__ emb_sub,
                        u16* __restrict__ xbf) {
  int idx = blockIdx.x * 256 + threadIdx.x;      // NPAD*128 total
  int n = idx >> 7, i = idx & 127;
  float v = 0.f;
  if (n < NN) {
    if (i < 16)       v = emb_type[x_type[n] * 16 + i];
    else if (i < 24)  v = emb_sub[x_sub[n] * 8 + (i - 16)];
    else if (i < 88)  v = x_bow[(size_t)n * 64 + (i - 24)];
    else if (i < 98)  v = x_hot[(size_t)n * 10 + (i - 88)];
    else if (i < 101) v = x_topo[(size_t)n * 3 + (i - 98)];
  }
  xbf[idx] = f2bf(v);
}

// ---------- 2: WMMA GEMM  h[rel] = x(bf16) @ W[rel]  (rel = blockIdx.y in [0,9)) ----------
__global__ void __launch_bounds__(256) k_gemm(const u16* __restrict__ xbf,
                                              const u16* __restrict__ wt,
                                              float* __restrict__ h,
                                              int kpad) {
  __shared__ __align__(16) u16 Wlds[64 * 128];   // transposed weights for this relation
  const int rel = blockIdx.y;
  const u16* wg = wt + (size_t)rel * 64 * kpad;
  const int tot4 = (64 * kpad) >> 3;             // 16-byte chunks

#if __has_builtin(__builtin_amdgcn_global_load_async_to_lds_b128)
  // Async DMA: global -> LDS directly (no VGPR round-trip), tracked by ASYNCcnt.
  {
    typedef int v4i __attribute__((vector_size(16)));
    __attribute__((address_space(3))) u16* ldsp =
        (__attribute__((address_space(3))) u16*)Wlds;
    __attribute__((address_space(1))) u16* gp =
        (__attribute__((address_space(1))) u16*)(u16*)wg;
    for (int i = threadIdx.x; i < tot4; i += 256)
      __builtin_amdgcn_global_load_async_to_lds_b128(
          (__attribute__((address_space(1))) v4i*)(gp + (size_t)i * 8),
          (__attribute__((address_space(3))) v4i*)(ldsp + (size_t)i * 8),
          0, 0);
    asm volatile("s_wait_asynccnt 0x0" ::: "memory");
  }
#else
  for (int i = threadIdx.x; i < tot4; i += 256)
    ((v4u*)Wlds)[i] = ((const v4u*)wg)[i];
#endif
  __syncthreads();

  const int wave  = threadIdx.x >> 5;
  const int lane  = threadIdx.x & 31;
  const int l15   = lane & 15;
  const int kb    = (lane < 16) ? 0 : 8;         // 16-bit A/B fragment K-halving
  const int baseM = blockIdx.x * 64 + (wave >> 1) * 16;
  const int baseN = (wave & 1) * 32;

  const u16* arow  = xbf  + (size_t)(baseM + l15) * kpad;
  const u16* brow0 = Wlds + (size_t)(baseN + l15) * kpad;
  const u16* brow1 = brow0 + (size_t)16 * kpad;

  __builtin_prefetch(arow, 0, 3);                // global_prefetch_b8

  v8f acc0 = {0.f, 0.f, 0.f, 0.f, 0.f, 0.f, 0.f, 0.f};
  v8f acc1 = {0.f, 0.f, 0.f, 0.f, 0.f, 0.f, 0.f, 0.f};

  for (int k0 = 0; k0 < kpad; k0 += 32) {
    v16bf a, b0, b1;
    // A 16x32: lane holds row baseM+l15, K = {k0+kb..+7, k0+16+kb..+7}
    *(v4u*)&a        = *(const v4u*)(arow + k0 + kb);
    *((v4u*)&a + 1)  = *(const v4u*)(arow + k0 + kb + 16);
    // B 32x16: lane holds col baseN+l15 (from transposed W), same K split
    *(v4u*)&b0       = *(const v4u*)(brow0 + k0 + kb);
    *((v4u*)&b0 + 1) = *(const v4u*)(brow0 + k0 + kb + 16);
    *(v4u*)&b1       = *(const v4u*)(brow1 + k0 + kb);
    *((v4u*)&b1 + 1) = *(const v4u*)(brow1 + k0 + kb + 16);
    acc0 = __builtin_amdgcn_wmma_f32_16x16x32_bf16(false, a, false, b0, (short)0, acc0, false, false);
    acc1 = __builtin_amdgcn_wmma_f32_16x16x32_bf16(false, a, false, b1, (short)0, acc1, false, false);
  }

  // D 16x16 f32 layout: VGPR v -> M = v + (lane>=16 ? 8 : 0), N = lane&15
  const int mOff = (lane >= 16) ? 8 : 0;
  float* out = h + ((size_t)rel * NPAD + baseM + mOff) * 64 + baseN + l15;
  #pragma unroll
  for (int v = 0; v < 8; ++v) {
    out[(size_t)v * 64]      = acc0[v];
    out[(size_t)v * 64 + 16] = acc1[v];
  }
}

// ---------- 3: edge scatter (segment sum + count per (rel,dst)) ----------
__global__ void k_agg(const float* __restrict__ h, const int* __restrict__ ei,
                      const int* __restrict__ et, float* __restrict__ sums,
                      float* __restrict__ cnt) {
  long long idx = (long long)blockIdx.x * 256 + threadIdx.x;   // E*64 total
  int e  = (int)(idx >> 6);
  int ch = (int)(idx & 63);
  int s = ei[e], d = ei[EE + e], r = et[e];
  float v = h[((size_t)r * NPAD + s) * 64 + ch];
  atomicAdd(&sums[((size_t)r * NPAD + d) * 64 + ch], v);
  if (ch == 0) atomicAdd(&cnt[(size_t)r * NPAD + d], 1.0f);
}

// ---------- 4: mean/combine + root + bias, accumulate BN statistics ----------
__global__ void k_combine(const float* __restrict__ sums, const float* __restrict__ cnt,
                          const float* __restrict__ h, const float* __restrict__ bias,
                          float* __restrict__ y, float* __restrict__ stats) {
  __shared__ float ssum[64], ssq[64];
  int tid = threadIdx.x;
  if (tid < 64) { ssum[tid] = 0.f; ssq[tid] = 0.f; }
  __syncthreads();
  size_t idx = (size_t)blockIdx.x * 256 + tid;    // N*64 total (exact multiple)
  int n = (int)(idx >> 6), ch = (int)(idx & 63);
  float acc = bias[ch];
  #pragma unroll
  for (int r = 0; r < RR; ++r) {
    float c  = cnt[(size_t)r * NPAD + n];
    float sv = sums[((size_t)r * NPAD + n) * 64 + ch];
    acc += sv / fmaxf(c, 1.0f);
  }
  acc += h[((size_t)RR * NPAD + n) * 64 + ch];    // x @ root (relation 8)
  y[idx] = acc;
  atomicAdd(&ssum[ch], acc);
  atomicAdd(&ssq[ch], acc * acc);
  __syncthreads();
  if (tid < 64) { atomicAdd(&stats[tid], ssum[tid]); atomicAdd(&stats[64 + tid], ssq[tid]); }
}

// ---------- 5: BN + ReLU, emit f32 and bf16 (stride 64) copies ----------
__global__ void k_bnapply(const float* __restrict__ y, const float* __restrict__ stats,
                          const float* __restrict__ gamma, const float* __restrict__ beta,
                          float* __restrict__ xf, u16* __restrict__ xbf) {
  size_t idx = (size_t)blockIdx.x * 256 + threadIdx.x;   // NPAD*64 total
  int n = (int)(idx >> 6), ch = (int)(idx & 63);
  float xv = 0.f;
  if (n < NN) {
    float mu  = stats[ch] * (1.0f / NN);
    float var = stats[64 + ch] * (1.0f / NN) - mu * mu;
    float t = (y[idx] - mu) * rsqrtf(var + 1e-5f) * gamma[ch] + beta[ch];
    xv = fmaxf(t, 0.f);
  }
  xf[idx]  = xv;
  xbf[idx] = f2bf(xv);
}

// ---------- 6: gate MLP (64->64 relu ->1) ----------
__global__ void __launch_bounds__(256) k_gate(const float* __restrict__ x,
                                              const float* __restrict__ w1, const float* __restrict__ b1,
                                              const float* __restrict__ w2, const float* __restrict__ b2,
                                              float* __restrict__ gate) {
  __shared__ float W[64 * 64];
  __shared__ float B1[64], W2[64];
  for (int i = threadIdx.x; i < 4096; i += 256) W[i] = w1[i];
  if (threadIdx.x < 64) { B1[threadIdx.x] = b1[threadIdx.x]; W2[threadIdx.x] = w2[threadIdx.x]; }
  __syncthreads();
  int n = blockIdx.x * 256 + threadIdx.x;
  if (n >= NN) return;
  const float* xr = x + (size_t)n * 64;
  float g = b2[0];
  for (int j = 0; j < 64; ++j) {
    float hj = B1[j];
    #pragma unroll 8
    for (int i = 0; i < 64; ++i) hj += xr[i] * W[i * 64 + j];
    g += fmaxf(hj, 0.f) * W2[j];
  }
  gate[n] = g;
}

// ---------- 7: segmented softmax pooling ----------
__global__ void k_gmax(const float* __restrict__ gate, const int* __restrict__ batch,
                       unsigned* __restrict__ gmaxe) {
  int n = blockIdx.x * 256 + threadIdx.x;
  if (n >= NN) return;
  atomicMax(&gmaxe[batch[n]], encf(gate[n]));
}
__global__ void k_exp(const float* __restrict__ gate, const int* __restrict__ batch,
                      const unsigned* __restrict__ gmaxe, float* __restrict__ egate,
                      float* __restrict__ denom) {
  int n = blockIdx.x * 256 + threadIdx.x;
  if (n >= NN) return;
  int b = batch[n];
  float e = expf(gate[n] - decf(gmaxe[b]));
  egate[n] = e;
  atomicAdd(&denom[b], e);
}
__global__ void k_pool(const float* __restrict__ egate, const float* __restrict__ denom,
                       const int* __restrict__ batch, const float* __restrict__ x,
                       float* __restrict__ gpool) {
  size_t idx = (size_t)blockIdx.x * 256 + threadIdx.x;   // N*64 total
  int n = (int)(idx >> 6), ch = (int)(idx & 63);
  int b = batch[n];
  float alpha = egate[n] / denom[b];
  atomicAdd(&gpool[(size_t)b * 64 + ch], alpha * x[idx]);
}
__global__ void k_heads(const float* __restrict__ gpool, const float* __restrict__ gfeats,
                        const float* __restrict__ ptw, const float* __restrict__ ptb,
                        const float* __restrict__ ciw, const float* __restrict__ cib,
                        float* __restrict__ out) {
  int g = threadIdx.x;                                   // 64 threads
  float apt = ptb[0], aci = cib[0];
  #pragma unroll
  for (int i = 0; i < 64; ++i) { float v = gpool[(size_t)g * 64 + i]; apt += v * ptw[i]; aci += v * ciw[i]; }
  #pragma unroll
  for (int i = 0; i < 12; ++i) { float v = gfeats[(size_t)g * 12 + i]; apt += v * ptw[64 + i]; aci += v * ciw[64 + i]; }
  out[g] = apt;
  out[64 + g] = aci;
}

// ---------- host ----------
extern "C" void kernel_launch(void* const* d_in, const int* in_sizes, int n_in,
                              void* d_out, int out_size, void* d_ws, size_t ws_size,
                              hipStream_t stream) {
  const int*   x_type   = (const int*)  d_in[0];
  const int*   x_sub    = (const int*)  d_in[1];
  const float* x_bow    = (const float*)d_in[2];
  const float* x_hot    = (const float*)d_in[3];
  const float* x_topo   = (const float*)d_in[4];
  const int*   eidx     = (const int*)  d_in[5];
  const int*   etype    = (const int*)  d_in[6];
  const int*   batch    = (const int*)  d_in[7];
  const float* g_feats  = (const float*)d_in[8];
  const float* emb_type = (const float*)d_in[9];
  const float* emb_sub  = (const float*)d_in[10];
  const float* bn_gamma = (const float*)d_in[11];
  const float* bn_beta  = (const float*)d_in[12];
  const float* gate_w1  = (const float*)d_in[13];
  const float* gate_b1  = (const float*)d_in[14];
  const float* gate_w2  = (const float*)d_in[15];
  const float* gate_b2  = (const float*)d_in[16];
  const float* head_pt_w = (const float*)d_in[17];
  const float* head_pt_b = (const float*)d_in[18];
  const float* head_ci_w = (const float*)d_in[19];
  const float* head_ci_b = (const float*)d_in[20];
  const float* comps[3]  = {(const float*)d_in[21], (const float*)d_in[25], (const float*)d_in[29]};
  const float* basess[3] = {(const float*)d_in[22], (const float*)d_in[26], (const float*)d_in[30]};
  const float* roots[3]  = {(const float*)d_in[23], (const float*)d_in[27], (const float*)d_in[31]};
  const float* biass[3]  = {(const float*)d_in[24], (const float*)d_in[28], (const float*)d_in[32]};

  size_t off = 0;
  char* base = (char*)d_ws;
  auto carve = [&](size_t bytes) -> void* {
    void* p = base + off;
    off += (bytes + 255) & ~(size_t)255;
    return p;
  };
  u16*      xbf   = (u16*)     carve((size_t)NPAD * 128 * 2);
  u16*      wt    = (u16*)     carve((size_t)9 * 64 * 128 * 2);
  float*    h     = (float*)   carve((size_t)9 * NPAD * 64 * 4);
  float*    sums  = (float*)   carve((size_t)RR * NPAD * 64 * 4);
  float*    cnt   = (float*)   carve((size_t)RR * NPAD * 4);
  float*    y     = (float*)   carve((size_t)NPAD * 64 * 4);
  float*    xf    = (float*)   carve((size_t)NPAD * 64 * 4);
  float*    stats = (float*)   carve(128 * 4);
  float*    gate  = (float*)   carve((size_t)NN * 4);
  float*    egate = (float*)   carve((size_t)NN * 4);
  unsigned* gmaxe = (unsigned*)carve(64 * 4);
  float*    denom = (float*)   carve(64 * 4);
  float*    gpool = (float*)   carve(64 * 64 * 4);
  (void)ws_size; (void)in_sizes; (void)n_in; (void)out_size;

  k_embed<<<(NPAD * 128) / 256, 256, 0, stream>>>(x_type, x_sub, x_bow, x_hot, x_topo,
                                                  emb_type, emb_sub, xbf);

  const int dins[3]  = {101, 64, 64};
  const int kpads[3] = {128, 64, 64};
  for (int l = 0; l < 3; ++l) {
    const int din = dins[l], kpad = kpads[l];
    k_wcombine<<<(9 * 64 * kpad) / 256, 256, 0, stream>>>(comps[l], basess[l], roots[l], wt, din, kpad);
    (void)hipMemsetAsync(sums,  0, (size_t)RR * NPAD * 64 * 4, stream);
    (void)hipMemsetAsync(cnt,   0, (size_t)RR * NPAD * 4, stream);
    (void)hipMemsetAsync(stats, 0, 128 * 4, stream);
    k_gemm<<<dim3(NPAD / 64, 9), 256, 0, stream>>>(xbf, wt, h, kpad);
    k_agg<<<(int)(((long long)EE * 64) / 256), 256, 0, stream>>>(h, eidx, etype, sums, cnt);
    k_combine<<<(NN * 64) / 256, 256, 0, stream>>>(sums, cnt, h, biass[l], y, stats);
    k_bnapply<<<(NPAD * 64) / 256, 256, 0, stream>>>(y, stats, bn_gamma, bn_beta, xf, xbf);
  }

  k_gate<<<(NN + 255) / 256, 256, 0, stream>>>(xf, gate_w1, gate_b1, gate_w2, gate_b2, gate);
  (void)hipMemsetAsync(gmaxe, 0, 64 * 4, stream);
  (void)hipMemsetAsync(denom, 0, 64 * 4, stream);
  (void)hipMemsetAsync(gpool, 0, 64 * 64 * 4, stream);
  k_gmax<<<(NN + 255) / 256, 256, 0, stream>>>(gate, batch, gmaxe);
  k_exp <<<(NN + 255) / 256, 256, 0, stream>>>(gate, batch, gmaxe, egate, denom);
  k_pool<<<(NN * 64) / 256, 256, 0, stream>>>(egate, denom, batch, xf, gpool);
  k_heads<<<1, 64, 0, stream>>>(gpool, g_feats, head_pt_w, head_pt_b, head_ci_w, head_ci_b,
                                (float*)d_out);
}
